// MultiHeadAttention_73607149519012
// MI455X (gfx1250) — matrix operand order repaired and checked by reading, not compiled, not run on hardware
//
#include <hip/hip_runtime.h>

typedef __attribute__((ext_vector_type(16))) __bf16 v16bf;
typedef __attribute__((ext_vector_type(8)))  float  v8f;

union BV { v16bf v; uint4 u[2]; };
union U2 { __bf16 h[4]; uint2 u; };

#define D_MODEL   1024
#define NUM_HEADS 16
#define DEPTH     64
#define BATCH     8
#define SEQ       1024

// ---- Tensor Data Mover availability (device pass only) -------------------
#if defined(__AMDGCN__) && __has_builtin(__builtin_amdgcn_tensor_load_to_lds) && __has_builtin(__builtin_amdgcn_s_wait_tensorcnt)
#define HAVE_TDM 1
#else
#define HAVE_TDM 0
#endif

#if HAVE_TDM
typedef unsigned int v4u __attribute__((ext_vector_type(4)));
typedef int          v8i __attribute__((ext_vector_type(8)));
typedef int          v4i __attribute__((ext_vector_type(4)));

// Issue a TDM load of a 32(keys) x 64(depth) bf16 tile, row stride D_MODEL,
// from global address gsrc into LDS byte offset lds_off. D# per ISA ch.8.
__device__ __forceinline__ void tdm_load_k(const __bf16* gsrc, unsigned lds_off)
{
    unsigned long long ga = (unsigned long long)(const void*)gsrc;
    v4u g0;
    g0[0] = 1u;                                        // count=1, user mode
    g0[1] = lds_off;                                   // lds_addr (bytes)
    g0[2] = (unsigned)(ga & 0xFFFFFFFFu);              // global_addr[31:0]
    g0[3] = (unsigned)((ga >> 32) & 0x01FFFFFFu) | (2u << 30);  // addr[56:32] | type=2
    v8i g1;
    g1[0] = (int)(1u << 16);                           // data_size=1 (2B), wg_mask=0
    g1[1] = (int)((unsigned)(D_MODEL & 0xFFFF) << 16); // tensor_dim0[15:0]
    g1[2] = (int)(((unsigned)D_MODEL >> 16) |
                  (((unsigned)(BATCH * SEQ) & 0xFFFF) << 16));   // dim0 hi | dim1 lo
    g1[3] = (int)(((unsigned)(BATCH * SEQ) >> 16) | (64u << 16)); // dim1 hi | tile_dim0=64
    g1[4] = (int)(32u);                                // tile_dim1=32, tile_dim2=0
    g1[5] = (int)(D_MODEL);                            // tensor_dim0_stride[31:0]
    g1[6] = 0;                                         // stride0 hi | stride1 lo
    g1[7] = 0;                                         // stride1 hi
    v4i gz = {0, 0, 0, 0};                             // 2D tensor: groups 2/3 unused
#if __clang_major__ >= 23
    v8i gz8 = {0, 0, 0, 0, 0, 0, 0, 0};
    __builtin_amdgcn_tensor_load_to_lds(g0, g1, gz, gz, gz8, 0);
#else
    __builtin_amdgcn_tensor_load_to_lds(g0, g1, gz, gz, 0);
#endif
}
#endif // HAVE_TDM

// -------------------------------------------------------------------------
// GEMM: out[M,N] = (A @ W + bias) * oscale
//   A: fp32 (CONCAT=0) or fp32 for k<KA0 then bf16 A1 for k>=KA0 (CONCAT=1).
//   W: fp32 row-major [K][N].  out: bf16 (OUT_F32=0) or fp32 (OUT_F32=1).
// Block tile 64x64, K-step 32, 4 waves each computing a 32x32 quadrant as
// 2x2 v_wmma_f32_16x16x32_bf16 tiles. Software-pipelined: global loads for
// tile k+1 are issued before the WMMAs of tile k. (fp32->bf16 conversion in
// flight rules out TDM here.)
// -------------------------------------------------------------------------
template<int CONCAT, int OUT_F32>
__global__ __launch_bounds__(128)
void gemm_bias_kernel(const float* __restrict__ A0, const __bf16* __restrict__ A1,
                      const float* __restrict__ W, const float* __restrict__ bias,
                      void* __restrict__ outp, int M, int N, int K, int KA0,
                      float oscale)
{
    (void)M;
    __shared__ __align__(16) __bf16 As[64 * 32];   // [m][k]
    __shared__ __align__(16) __bf16 Bs[64 * 32];   // [n][k] (transposed)

    const int t  = threadIdx.x;
    const int m0 = blockIdx.x * 64;
    const int n0 = blockIdx.y * 64;
    const int w  = t >> 5;
    const int lr = t & 15;              // lane % 16
    const int hi = (t >> 4) & 1;        // lane >= 16 within wave
    const int mq = (w >> 1) * 32;
    const int nq = (w & 1) * 32;

    v8f acc00 = {}, acc01 = {}, acc10 = {}, acc11 = {};

    float4 a_f[4]; uint2 a_b[4]; float4 w_f[4];

    auto load_tile = [&](int kk) {
        #pragma unroll
        for (int i = 0; i < 4; ++i) {
            int s   = t + i * 128;              // 0..511
            int row = s >> 3;                   // A row 0..63
            int kq  = (s & 7) << 2;             // A k 0,4,..28
            int gk  = kk + kq;
            if (CONCAT && kk >= KA0)
                a_b[i] = *(const uint2*)(A1 + (size_t)(m0 + row) * (size_t)(K - KA0) + (gk - KA0));
            else
                a_f[i] = *(const float4*)(A0 + (size_t)(m0 + row) * KA0 + gk);
            int kb  = s >> 4;                   // W k 0..31
            int nqq = (s & 15) << 2;            // W n 0,4,..60
            w_f[i] = *(const float4*)(W + (size_t)(kk + kb) * N + n0 + nqq);
        }
    };
    auto store_tile = [&](int kk) {
        #pragma unroll
        for (int i = 0; i < 4; ++i) {
            int s   = t + i * 128;
            int row = s >> 3;
            int kq  = (s & 7) << 2;
            U2 pk;
            if (CONCAT && kk >= KA0) {
                pk.u = a_b[i];
            } else {
                pk.h[0] = (__bf16)a_f[i].x; pk.h[1] = (__bf16)a_f[i].y;
                pk.h[2] = (__bf16)a_f[i].z; pk.h[3] = (__bf16)a_f[i].w;
            }
            *(uint2*)&As[row * 32 + kq] = pk.u;
            int kb  = s >> 4;
            int nqq = (s & 15) << 2;
            Bs[(nqq + 0) * 32 + kb] = (__bf16)w_f[i].x;
            Bs[(nqq + 1) * 32 + kb] = (__bf16)w_f[i].y;
            Bs[(nqq + 2) * 32 + kb] = (__bf16)w_f[i].z;
            Bs[(nqq + 3) * 32 + kb] = (__bf16)w_f[i].w;
        }
    };

    load_tile(0);
    for (int kk = 0; kk < K; kk += 32) {
        __syncthreads();                 // previous tile's LDS reads complete
        store_tile(kk);
        __syncthreads();                 // staging visible to all waves
        if (kk + 32 < K) load_tile(kk + 32);   // next tile in flight behind WMMAs

        BV a0, a1, b0, b1;
        const __bf16* ab0 = &As[(mq +  0 + lr) * 32] + (hi ? 8 : 0);
        a0.u[0] = *(const uint4*)(ab0);
        a0.u[1] = *(const uint4*)(ab0 + 16);
        const __bf16* ab1 = &As[(mq + 16 + lr) * 32] + (hi ? 8 : 0);
        a1.u[0] = *(const uint4*)(ab1);
        a1.u[1] = *(const uint4*)(ab1 + 16);
        const __bf16* bb0 = &Bs[(nq +  0 + lr) * 32 + (hi ? 8 : 0)];
        b0.u[0] = *(const uint4*)(bb0);
        b0.u[1] = *(const uint4*)(bb0 + 16);
        const __bf16* bb1 = &Bs[(nq + 16 + lr) * 32 + (hi ? 8 : 0)];
        b1.u[0] = *(const uint4*)(bb1);
        b1.u[1] = *(const uint4*)(bb1 + 16);

        acc00 = __builtin_amdgcn_wmma_f32_16x16x32_bf16(false, a0.v, false, b0.v, (short)0, acc00, false, false);
        acc01 = __builtin_amdgcn_wmma_f32_16x16x32_bf16(false, a0.v, false, b1.v, (short)0, acc01, false, false);
        acc10 = __builtin_amdgcn_wmma_f32_16x16x32_bf16(false, a1.v, false, b0.v, (short)0, acc10, false, false);
        acc11 = __builtin_amdgcn_wmma_f32_16x16x32_bf16(false, a1.v, false, b1.v, (short)0, acc11, false, false);
    }

    // ---- epilogue: bias, scale, store (C layout: lane = col, vgpr = row) ----
    #pragma unroll
    for (int s = 0; s < 2; ++s) {
        #pragma unroll
        for (int tt = 0; tt < 2; ++tt) {
            v8f acc = (s == 0) ? (tt == 0 ? acc00 : acc01)
                               : (tt == 0 ? acc10 : acc11);
            int col = n0 + nq + tt * 16 + lr;
            float bvv = bias[col];
            #pragma unroll
            for (int j = 0; j < 8; ++j) {
                int row = m0 + mq + s * 16 + j + hi * 8;
                float vo = (acc[j] + bvv) * oscale;
                if (OUT_F32) ((float*)outp)[(size_t)row * N + col] = vo;
                else         ((__bf16*)outp)[(size_t)row * N + col] = (__bf16)vo;
            }
        }
    }
}

// -------------------------------------------------------------------------
// Flash attention: block = (batch b, head h, 64 queries); 4 waves, each a
// 16-query tile. Q pre-scaled by 1/sqrt(depth). Online softmax; row max via
// half-wave ds_bpermute reduction, row sum via an extra P @ ones WMMA.
// K chunk staged by the Tensor Data Mover into double-buffered LDS (wave 0
// issues, s_wait_tensorcnt + barrier publishes); V register-pipelined into
// a transposed LDS tile (TDM cannot transpose).
// -------------------------------------------------------------------------
__global__ __launch_bounds__(128)
void attn_kernel(const __bf16* __restrict__ Qw, const __bf16* __restrict__ Kw,
                 const __bf16* __restrict__ Vw, const float* __restrict__ mask,
                 __bf16* __restrict__ Cw)
{
    // Single block so TDM lds_addr offsets are compile-time constants:
    //   Ks buf0 @ byte 0, Ks buf1 @ 4096, Vs @ 8192, Ps @ 12288
    __shared__ __align__(16) __bf16 SMEM[2 * 2048 + 2048 + 4 * 512];
    __bf16* const Vs = SMEM + 4096;                  // [d][key] 64x32
    __bf16* const Ps = SMEM + 6144;                  // per-wave 16x32 tiles

    const int t  = threadIdx.x;
    const int b  = blockIdx.x >> 4;
    const int h  = blockIdx.x & 15;
    const int w  = t >> 5;
    const int lr = t & 15;
    const int hi = (t >> 4) & 1;
    const int q0 = blockIdx.y * 64 + w * 16;

    // Q tile held in registers as two K=32 A-operands (depth 0..31, 32..63)
    const __bf16* qrow = Qw + (size_t)(b * SEQ + q0 + lr) * D_MODEL + h * DEPTH + (hi ? 8 : 0);
    BV aq0, aq1;
    aq0.u[0] = *(const uint4*)(qrow);
    aq0.u[1] = *(const uint4*)(qrow + 16);
    aq1.u[0] = *(const uint4*)(qrow + 32);
    aq1.u[1] = *(const uint4*)(qrow + 48);

    // constant all-ones B operand for row-sum WMMA
    BV ones;
    #pragma unroll
    for (int i = 0; i < 16; ++i) ones.v[i] = (__bf16)1.0f;

    float mrow[8], ssum[8];
    #pragma unroll
    for (int j = 0; j < 8; ++j) { mrow[j] = -3.0e38f; ssum[j] = 0.0f; }
    v8f c0 = {}, c1 = {}, c2 = {}, c3 = {};

    uint2 vreg[4];
#if !HAVE_TDM
    uint2 kreg[4];
#endif
    auto load_chunk = [&](int kc) {
        #pragma unroll
        for (int i = 0; i < 4; ++i) {
            int s   = t + i * 128;
            int key = s >> 4;                   // 0..31
            int dq  = (s & 15) << 2;            // 0,4,..60
            size_t base = (size_t)(b * SEQ + kc + key) * D_MODEL + h * DEPTH + dq;
            vreg[i] = *(const uint2*)(Vw + base);
#if !HAVE_TDM
            kreg[i] = *(const uint2*)(Kw + base);
#endif
        }
    };
    auto store_chunk = [&](__bf16* Kcur) {
        #pragma unroll
        for (int i = 0; i < 4; ++i) {
            int s   = t + i * 128;
            int key = s >> 4;
            int dq  = (s & 15) << 2;
            U2 vu; vu.u = vreg[i];
            Vs[(dq + 0) * 32 + key] = vu.h[0];
            Vs[(dq + 1) * 32 + key] = vu.h[1];
            Vs[(dq + 2) * 32 + key] = vu.h[2];
            Vs[(dq + 3) * 32 + key] = vu.h[3];
#if !HAVE_TDM
            *(uint2*)&Kcur[key * 64 + dq] = kreg[i];
#else
            (void)Kcur;
#endif
        }
    };

#if HAVE_TDM
    if (w == 0)                                   // wave-uniform branch: TDM ignores EXEC
        tdm_load_k(Kw + (size_t)(b * SEQ) * D_MODEL + h * DEPTH, 0u);
#endif
    load_chunk(0);

    int buf = 0;
    for (int kc = 0; kc < SEQ; kc += 32) {
        __bf16* const Kcur = SMEM + buf * 2048;   // [key][d] 32x64

        __syncthreads();                  // previous chunk's Ks/Vs reads done
        store_chunk(Kcur);
#if HAVE_TDM
        if (w == 0) {                     // overlap next TDM behind this wait
            if (kc + 32 < SEQ) {
                tdm_load_k(Kw + (size_t)(b * SEQ + kc + 32) * D_MODEL + h * DEPTH,
                           (unsigned)((buf ^ 1) * 4096));
                __builtin_amdgcn_s_wait_tensorcnt(1);
            } else {
                __builtin_amdgcn_s_wait_tensorcnt(0);
            }
        }
#endif
        __syncthreads();                  // staging (TDM K + V) visible
        if (kc + 32 < SEQ) load_chunk(kc + 32);   // next V (and K fallback) behind compute

        // ---- logits = Q @ K^T for 2 key subtiles, depth in 2 K=32 halves ----
        v8f lg0 = {}, lg1 = {};
        #pragma unroll
        for (int half = 0; half < 2; ++half) {
            BV bk0, bk1;
            const __bf16* kb0 = &Kcur[( 0 + lr) * 64 + half * 32 + (hi ? 8 : 0)];
            bk0.u[0] = *(const uint4*)(kb0);
            bk0.u[1] = *(const uint4*)(kb0 + 16);
            const __bf16* kb1 = &Kcur[(16 + lr) * 64 + half * 32 + (hi ? 8 : 0)];
            bk1.u[0] = *(const uint4*)(kb1);
            bk1.u[1] = *(const uint4*)(kb1 + 16);
            v16bf aqh = (half == 0) ? aq0.v : aq1.v;
            lg0 = __builtin_amdgcn_wmma_f32_16x16x32_bf16(false, aqh, false, bk0.v, (short)0, lg0, false, false);
            lg1 = __builtin_amdgcn_wmma_f32_16x16x32_bf16(false, aqh, false, bk1.v, (short)0, lg1, false, false);
        }

        // additive mask (zeros for this problem; kept for fidelity)
        float mk0 = mask[b * SEQ + kc + lr]      * -1e9f;
        float mk1 = mask[b * SEQ + kc + 16 + lr] * -1e9f;
        #pragma unroll
        for (int j = 0; j < 8; ++j) { lg0[j] += mk0; lg1[j] += mk1; }

        // ---- row max over the 16 lanes holding each row, then p = exp ----
        float corr[8];
        #pragma unroll
        for (int j = 0; j < 8; ++j) {
            float rm = fmaxf(lg0[j], lg1[j]);
            rm = fmaxf(rm, __shfl_xor(rm, 1, 32));
            rm = fmaxf(rm, __shfl_xor(rm, 2, 32));
            rm = fmaxf(rm, __shfl_xor(rm, 4, 32));
            rm = fmaxf(rm, __shfl_xor(rm, 8, 32));
            float nm = fmaxf(mrow[j], rm);
            corr[j] = __expf(mrow[j] - nm);
            mrow[j] = nm;
            lg0[j] = __expf(lg0[j] - nm);
            lg1[j] = __expf(lg1[j] - nm);
        }

        // ---- P (C layout) -> per-wave LDS -> A-operand layout ----
        __bf16* pw = Ps + w * 512;
        #pragma unroll
        for (int j = 0; j < 8; ++j) {
            int row = j + hi * 8;
            pw[row * 32 + lr]      = (__bf16)lg0[j];
            pw[row * 32 + 16 + lr] = (__bf16)lg1[j];
        }
        // wave-private LDS round trip: hardware keeps DS ops in order per
        // wave; only compiler reordering must be prevented.
        asm volatile("" ::: "memory");

        BV ap;
        const __bf16* pb = pw + lr * 32 + (hi ? 8 : 0);
        ap.u[0] = *(const uint4*)(pb);
        ap.u[1] = *(const uint4*)(pb + 16);

        // ---- row sums via P @ ones (replaces 32 ds_bpermute reductions) ----
        v8f rsv = {};
        rsv = __builtin_amdgcn_wmma_f32_16x16x32_bf16(false, ap.v, false, ones.v, (short)0, rsv, false, false);

        #pragma unroll
        for (int j = 0; j < 8; ++j) {
            ssum[j] = ssum[j] * corr[j] + rsv[j];
            c0[j] *= corr[j]; c1[j] *= corr[j];
            c2[j] *= corr[j]; c3[j] *= corr[j];
        }

        // ---- ctx += P @ V over 4 depth subtiles ----
        #pragma unroll
        for (int t2 = 0; t2 < 4; ++t2) {
            BV bvv;
            const __bf16* vb = &Vs[(t2 * 16 + lr) * 32 + (hi ? 8 : 0)];
            bvv.u[0] = *(const uint4*)(vb);
            bvv.u[1] = *(const uint4*)(vb + 16);
            if      (t2 == 0) c0 = __builtin_amdgcn_wmma_f32_16x16x32_bf16(false, ap.v, false, bvv.v, (short)0, c0, false, false);
            else if (t2 == 1) c1 = __builtin_amdgcn_wmma_f32_16x16x32_bf16(false, ap.v, false, bvv.v, (short)0, c1, false, false);
            else if (t2 == 2) c2 = __builtin_amdgcn_wmma_f32_16x16x32_bf16(false, ap.v, false, bvv.v, (short)0, c2, false, false);
            else              c3 = __builtin_amdgcn_wmma_f32_16x16x32_bf16(false, ap.v, false, bvv.v, (short)0, c3, false, false);
        }
        buf ^= 1;
    }

    // ---- normalize and store ctx (bf16, (B,S,H*depth) layout) ----
    #pragma unroll
    for (int j = 0; j < 8; ++j) {
        float inv = 1.0f / ssum[j];
        size_t row = (size_t)(b * SEQ + q0 + j + hi * 8) * D_MODEL + h * DEPTH;
        Cw[row + lr]      = (__bf16)(c0[j] * inv);
        Cw[row + 16 + lr] = (__bf16)(c1[j] * inv);
        Cw[row + 32 + lr] = (__bf16)(c2[j] * inv);
        Cw[row + 48 + lr] = (__bf16)(c3[j] * inv);
    }
}

// -------------------------------------------------------------------------
extern "C" void kernel_launch(void* const* d_in, const int* in_sizes, int n_in,
                              void* d_out, int out_size, void* d_ws, size_t ws_size,
                              hipStream_t stream)
{
    (void)in_sizes; (void)n_in; (void)out_size; (void)ws_size;
    const float* v    = (const float*)d_in[0];
    const float* k    = (const float*)d_in[1];
    const float* q_in = (const float*)d_in[2];
    const float* mask = (const float*)d_in[3];
    const float* wq_w = (const float*)d_in[4];
    const float* wq_b = (const float*)d_in[5];
    const float* wk_w = (const float*)d_in[6];
    const float* wk_b = (const float*)d_in[7];
    const float* wv_w = (const float*)d_in[8];
    const float* wv_b = (const float*)d_in[9];
    const float* wo_w = (const float*)d_in[10];
    const float* wo_b = (const float*)d_in[11];
    float* out = (float*)d_out;

    const size_t MT = (size_t)BATCH * SEQ;         // 8192
    __bf16* Qws = (__bf16*)d_ws;
    __bf16* Kws = Qws + MT * D_MODEL;
    __bf16* Vws = Kws + MT * D_MODEL;
    __bf16* Cws = Vws + MT * D_MODEL;              // 64 MB total bf16 scratch

    dim3 bb(128);
    dim3 gg((unsigned)(MT / 64), D_MODEL / 64);    // (128, 16)
    const float qscale = 0.125f;                   // 1/sqrt(DEPTH)

    // Q/K/V projections (Q pre-scaled so attention skips logits scaling)
    gemm_bias_kernel<0, 0><<<gg, bb, 0, stream>>>(q_in, (const __bf16*)nullptr, wq_w, wq_b, Qws, (int)MT, D_MODEL, D_MODEL, D_MODEL, qscale);
    gemm_bias_kernel<0, 0><<<gg, bb, 0, stream>>>(k,    (const __bf16*)nullptr, wk_w, wk_b, Kws, (int)MT, D_MODEL, D_MODEL, D_MODEL, 1.0f);
    gemm_bias_kernel<0, 0><<<gg, bb, 0, stream>>>(v,    (const __bf16*)nullptr, wv_w, wv_b, Vws, (int)MT, D_MODEL, D_MODEL, D_MODEL, 1.0f);

    // Flash attention
    dim3 ga(BATCH * NUM_HEADS, SEQ / 64);          // (128, 16)
    attn_kernel<<<ga, bb, 0, stream>>>(Qws, Kws, Vws, mask, Cws);

    // Output projection over concat([q_in, ctx]) without materializing it
    gemm_bias_kernel<1, 1><<<gg, bb, 0, stream>>>(q_in, Cws, wo_w, wo_b, out, (int)MT, D_MODEL, 2 * D_MODEL, D_MODEL, 1.0f);
}
